// EAGLE_36275293782545
// MI455X (gfx1250) — compile-verified, matrix operand order checked by reading, not run on hardware
//
#include <hip/hip_runtime.h>

typedef __attribute__((ext_vector_type(16))) __bf16 v16bf;
typedef __attribute__((ext_vector_type(8)))  float  v8f;

constexpr int BATCH = 32;
constexpr int NNODE = 512;
constexpr int DIN   = 768;
constexpr int DOUT  = 256;
constexpr float SLOPE = 0.2f;
constexpr float NEGF  = -9.0e15f;

constexpr int TM   = 128;  // output rows per workgroup
constexpr int KC   = 32;   // K per WMMA step (bf16 16x16x32)
constexpr int APAD = 40;   // padded LDS row stride in bf16 elems (conflict-free b128)

// Load one 16x16x32 bf16 A/B fragment half-row from LDS.
// Lane (l, hi) holds K = {hi*8 .. hi*8+7} and {16+hi*8 .. 16+hi*8+7}.
__device__ __forceinline__ v16bf ldsFrag(const __bf16* rowBase, int hi) {
  union { uint4 u[2]; v16bf v; } f;
  f.u[0] = *reinterpret_cast<const uint4*>(rowBase + hi * 8);
  f.u[1] = *reinterpret_cast<const uint4*>(rowBase + 16 + hi * 8);
  return f.v;
}

// Async global -> LDS copy of 16 bytes (ASYNCcnt-tracked, bypasses VGPRs).
// GV addressing: 64-bit global address in a VGPR pair, LDS byte offset in VGPR.
__device__ __forceinline__ void asyncCopyB128(unsigned lds_off, const void* gptr) {
  asm volatile("global_load_async_to_lds_b128 %0, %1, off"
               :: "v"(lds_off), "v"((unsigned long long)(uintptr_t)gptr)
               : "memory");
}

// ---------------- W transpose + bf16 convert: Wt[d][k] = bf16(W[k][d]) ----
__global__ __launch_bounds__(256) void gat_wt(const float* __restrict__ W,
                                              __bf16* __restrict__ Wt) {
  const int idx = blockIdx.x * 256 + threadIdx.x;   // over DIN*DOUT
  const int k = idx >> 8;        // / DOUT
  const int d = idx & (DOUT - 1);
  Wt[(size_t)d * DIN + k] = (__bf16)W[idx];
}

// ---------------- GEMM1: Wh = h @ W + pos_table[positions] ---------------
__global__ __launch_bounds__(256) void gat_gemm1(
    const float* __restrict__ h, const __bf16* __restrict__ Wt,
    const int* __restrict__ positions, const float* __restrict__ pos_table,
    float* __restrict__ Wh, __bf16* __restrict__ WhT) {
  __shared__ alignas(16) __bf16 sA[TM * APAD];    // 128 rows x 32 k (padded)
  __shared__ alignas(16) __bf16 sB[DOUT * APAD];  // 256 d-rows x 32 k (padded)
  const int tid  = threadIdx.x;
  const int wave = tid >> 5;
  const int lane = tid & 31;
  const int l    = lane & 15;
  const int hi   = lane >> 4;
  const int rbase = blockIdx.x * TM;

  const v8f vzero = {0.f, 0.f, 0.f, 0.f, 0.f, 0.f, 0.f, 0.f};
  v8f acc[16];
#pragma unroll
  for (int i = 0; i < 16; ++i) acc[i] = vzero;

  for (int k0 = 0; k0 < DIN; k0 += KC) {
    __syncthreads();
    {  // stage A: f32 -> bf16, 2 threads per row (16 elems each)
      const int r = tid >> 1, half = tid & 1;
      const float4* src = reinterpret_cast<const float4*>(
          h + (size_t)(rbase + r) * DIN + k0 + half * 16);
      const float4 q0 = src[0], q1 = src[1], q2 = src[2], q3 = src[3];
      const float fs[16] = {q0.x, q0.y, q0.z, q0.w, q1.x, q1.y, q1.z, q1.w,
                            q2.x, q2.y, q2.z, q2.w, q3.x, q3.y, q3.z, q3.w};
      union { __bf16 b[16]; uint4 u[2]; } cv;
#pragma unroll
      for (int i = 0; i < 16; ++i) cv.b[i] = (__bf16)fs[i];
      uint4* dst = reinterpret_cast<uint4*>(sA + r * APAD + half * 16);
      dst[0] = cv.u[0];
      dst[1] = cv.u[1];
    }
    {  // stage B: bf16 copy, 1 thread per d-row (32 elems = 64B)
      const uint4* src =
          reinterpret_cast<const uint4*>(Wt + (size_t)tid * DIN + k0);
      const uint4 u0 = src[0], u1 = src[1], u2 = src[2], u3 = src[3];
      uint4* dst = reinterpret_cast<uint4*>(sB + tid * APAD);
      dst[0] = u0; dst[1] = u1; dst[2] = u2; dst[3] = u3;
    }
    __syncthreads();
    const v16bf afrag = ldsFrag(sA + (wave * 16 + l) * APAD, hi);
#pragma unroll
    for (int nt = 0; nt < 16; ++nt) {
      const v16bf bfrag = ldsFrag(sB + (nt * 16 + l) * APAD, hi);
      acc[nt] = __builtin_amdgcn_wmma_f32_16x16x32_bf16(
          false, afrag, false, bfrag, (short)0, acc[nt], false, false);
    }
  }
  // Epilogue: C lane holds M = r + 8*hi, N = l per tile nt.
  const int row0 = rbase + wave * 16 + hi * 8;
#pragma unroll
  for (int r = 0; r < 8; ++r) {
    const int row = row0 + r;
    const int pos = positions[row];
    const float* pt = pos_table + (size_t)pos * DOUT;
    const int bb = row / NNODE;
    const int ii = row % NNODE;
#pragma unroll
    for (int nt = 0; nt < 16; ++nt) {
      const int col = nt * 16 + l;
      const float v = acc[nt][r] + pt[col];
      Wh[(size_t)row * DOUT + col] = v;
      WhT[((size_t)bb * DOUT + col) * NNODE + ii] = (__bf16)v;
    }
  }
}

// ---------------- s1/s2: per-row dots with a1, a2 ------------------------
__global__ __launch_bounds__(256) void gat_scores(
    const float* __restrict__ Wh, const float* __restrict__ a1,
    const float* __restrict__ a2, float* __restrict__ s1,
    float* __restrict__ s2) {
  const int lane = threadIdx.x & 31;
  const int row  = blockIdx.x * 8 + (threadIdx.x >> 5);
  const float* wr = Wh + (size_t)row * DOUT;
  float x1 = 0.f, x2 = 0.f;
#pragma unroll
  for (int t = 0; t < 8; ++t) {
    const int d = lane + t * 32;
    const float w = wr[d];
    x1 += w * a1[d];
    x2 += w * a2[d];
  }
#pragma unroll
  for (int off = 16; off > 0; off >>= 1) {
    x1 += __shfl_xor(x1, off, 32);
    x2 += __shfl_xor(x2, off, 32);
  }
  if (lane == 0) { s1[row] = x1; s2[row] = x2; }
}

// ---------------- masked leakyrelu + softmax row -------------------------
__global__ __launch_bounds__(256) void gat_attention(
    const float* __restrict__ s1, const float* __restrict__ s2,
    const int* __restrict__ adj, float* __restrict__ attnF,
    __bf16* __restrict__ attnB) {
  const int lane = threadIdx.x & 31;
  const int row  = blockIdx.x * 8 + (threadIdx.x >> 5);
  const int bb   = row >> 9;  // / NNODE
  const float sv = s1[row];
  const float* s2b = s2 + (size_t)bb * NNODE;
  const int* ar = adj + (size_t)row * NNODE;
  float vals[16];
  float mx = NEGF;
#pragma unroll
  for (int t = 0; t < 16; ++t) {
    const int j = lane + t * 32;
    float e = sv + s2b[j];
    e = (e > 0.f) ? e : SLOPE * e;
    e = (ar[j] > 0) ? e : NEGF;
    vals[t] = e;
    mx = fmaxf(mx, e);
  }
#pragma unroll
  for (int off = 16; off > 0; off >>= 1) mx = fmaxf(mx, __shfl_xor(mx, off, 32));
  float sum = 0.f;
#pragma unroll
  for (int t = 0; t < 16; ++t) { vals[t] = __expf(vals[t] - mx); sum += vals[t]; }
#pragma unroll
  for (int off = 16; off > 0; off >>= 1) sum += __shfl_xor(sum, off, 32);
  const float inv = 1.0f / sum;
  float*  of = attnF + (size_t)row * NNODE;
  __bf16* ob = attnB + (size_t)row * NNODE;
#pragma unroll
  for (int t = 0; t < 16; ++t) {
    const int j = lane + t * 32;
    const float p = vals[t] * inv;
    of[j] = p;
    ob[j] = (__bf16)p;
  }
}

// ---------------- GEMM2: h' = attn @ Wh (batched) ------------------------
// Double-buffered: K-chunk k+1 streamed into LDS with async-to-LDS loads
// while chunk k is consumed by WMMA; ASYNCcnt + barrier for handoff.
__global__ __launch_bounds__(256) void gat_gemm2(
    const __bf16* __restrict__ attnB, const __bf16* __restrict__ WhT,
    float* __restrict__ hprime) {
  __shared__ alignas(16) __bf16 sA[2][TM * APAD];
  __shared__ alignas(16) __bf16 sB[2][DOUT * APAD];
  const int tid  = threadIdx.x;
  const int wave = tid >> 5;
  const int lane = tid & 31;
  const int l    = lane & 15;
  const int hi   = lane >> 4;
  const int rbase = blockIdx.x * TM;     // 128 rows, never straddles a batch
  const int bb    = rbase / NNODE;

  // Per-thread staging coordinates:
  //  A: 2 threads per attention row, 32B (2 x b128) each
  //  B: 1 thread per d-row of WhT, 64B (4 x b128) each
  const int ar = tid >> 1, ah = tid & 1;
  const __bf16* agsrc = attnB + (size_t)(rbase + ar) * NNODE + ah * 16;
  const __bf16* bgsrc = WhT + ((size_t)bb * DOUT + tid) * NNODE;
  const unsigned aoff[2] = {
      (unsigned)(uintptr_t)&sA[0][ar * APAD + ah * 16],
      (unsigned)(uintptr_t)&sA[1][ar * APAD + ah * 16]};
  const unsigned boff[2] = {
      (unsigned)(uintptr_t)&sB[0][tid * APAD],
      (unsigned)(uintptr_t)&sB[1][tid * APAD]};

  const v8f vzero = {0.f, 0.f, 0.f, 0.f, 0.f, 0.f, 0.f, 0.f};
  v8f acc[16];
#pragma unroll
  for (int i = 0; i < 16; ++i) acc[i] = vzero;

  auto issueChunk = [&](int k0, int buf) {
    const __bf16* ga = agsrc + k0;
    asyncCopyB128(aoff[buf], ga);
    asyncCopyB128(aoff[buf] + 16, ga + 8);
    const __bf16* gb = bgsrc + k0;
    asyncCopyB128(boff[buf], gb);
    asyncCopyB128(boff[buf] + 16, gb + 8);
    asyncCopyB128(boff[buf] + 32, gb + 16);
    asyncCopyB128(boff[buf] + 48, gb + 24);
  };

  constexpr int NK = NNODE / KC;  // 16 K-chunks
  issueChunk(0, 0);
  for (int k = 0; k < NK; ++k) {
    const int buf = k & 1;
    if (k + 1 < NK) {
      issueChunk((k + 1) * KC, buf ^ 1);
      // 6 newer ops outstanding for chunk k+1; async loads complete in
      // order, so asynccnt <= 6 means chunk k's data is in LDS.
      asm volatile("s_wait_asynccnt 0x6" ::: "memory");
    } else {
      asm volatile("s_wait_asynccnt 0x0" ::: "memory");
    }
    __syncthreads();  // chunk-k data visible to all waves
    const v16bf afrag = ldsFrag(&sA[buf][(wave * 16 + l) * APAD], hi);
#pragma unroll
    for (int nt = 0; nt < 16; ++nt) {
      const v16bf bfrag = ldsFrag(&sB[buf][(nt * 16 + l) * APAD], hi);
      acc[nt] = __builtin_amdgcn_wmma_f32_16x16x32_bf16(
          false, afrag, false, bfrag, (short)0, acc[nt], false, false);
    }
    __syncthreads();  // everyone done reading buf before it is refilled
  }
  const int row0 = rbase + wave * 16 + hi * 8;
#pragma unroll
  for (int r = 0; r < 8; ++r) {
    const int row = row0 + r;
#pragma unroll
    for (int nt = 0; nt < 16; ++nt) {
      hprime[(size_t)row * DOUT + nt * 16 + l] = acc[nt][r];
    }
  }
}

extern "C" void kernel_launch(void* const* d_in, const int* in_sizes, int n_in,
                              void* d_out, int out_size, void* d_ws,
                              size_t ws_size, hipStream_t stream) {
  (void)in_sizes; (void)n_in; (void)out_size; (void)ws_size;
  const float* h    = (const float*)d_in[0];
  const int*   adj  = (const int*)d_in[1];
  const int*   pos  = (const int*)d_in[2];
  const float* W    = (const float*)d_in[3];
  const float* a1   = (const float*)d_in[4];
  const float* a2   = (const float*)d_in[5];
  const float* ptab = (const float*)d_in[6];

  float* hprime = (float*)d_out;                                   // [B,N,DOUT]
  float* attnF  = hprime + (size_t)BATCH * NNODE * DOUT;           // [B,N,N]

  char* p = (char*)d_ws;
  float*  Wh    = (float*)p;  p += (size_t)BATCH * NNODE * DOUT * 4;
  __bf16* WhT   = (__bf16*)p; p += (size_t)BATCH * DOUT * NNODE * 2;
  __bf16* Wt    = (__bf16*)p; p += (size_t)DOUT * DIN * 2;
  __bf16* attnB = (__bf16*)p; p += (size_t)BATCH * NNODE * NNODE * 2;
  float*  s1    = (float*)p;  p += (size_t)BATCH * NNODE * 4;
  float*  s2    = (float*)p;  p += (size_t)BATCH * NNODE * 4;

  gat_wt<<<(DIN * DOUT) / 256, 256, 0, stream>>>(W, Wt);
  gat_gemm1<<<(BATCH * NNODE) / TM, 256, 0, stream>>>(h, Wt, pos, ptab, Wh, WhT);
  gat_scores<<<(BATCH * NNODE) / 8, 256, 0, stream>>>(Wh, a1, a2, s1, s2);
  gat_attention<<<(BATCH * NNODE) / 8, 256, 0, stream>>>(s1, s2, adj, attnF, attnB);
  gat_gemm2<<<(BATCH * NNODE) / TM, 256, 0, stream>>>(attnB, WhT, hprime);
}
